// EGNN_47828755808708
// MI455X (gfx1250) — matrix operand order repaired and checked by reading, not compiled, not run on hardware
//
#include <hip/hip_runtime.h>
#include <hip/hip_bf16.h>

#define N_NODES 50000
#define N_EDGES 800000
#define DH      128
#define ED_DIM  51
#define KP_E1   320   // 308 padded to 10 chunks of 32
#define KP_N1   256
#define EPS_R   1e-8f

typedef __attribute__((ext_vector_type(16))) __bf16 v16bf;
typedef __attribute__((ext_vector_type(8)))  __bf16 v8bf;
typedef __attribute__((ext_vector_type(8)))  float  v8f;
typedef __attribute__((ext_vector_type(4)))  float  f32x4;

__device__ __forceinline__ float silu_f(float x) { return x / (1.0f + __expf(-x)); }

__device__ __forceinline__ v8f wmma_bf16(v16bf a, v16bf b, v8f c) {
    return __builtin_amdgcn_wmma_f32_16x16x32_bf16(false, a, false, b, (short)0, c, false, false);
}

// -------------------- weight prep: f32 [Kin,Nout] -> bf16 transposed [Nout,KP] --------------------
__global__ void prep_wt_kernel(const float* __restrict__ W, __bf16* __restrict__ Wt,
                               int Kin, int Nout, int KP) {
    int idx = blockIdx.x * blockDim.x + threadIdx.x;
    if (idx >= Nout * KP) return;
    int n = idx / KP;
    int k = idx - n * KP;
    float v = (k < Kin) ? W[(size_t)k * Nout + n] : 0.0f;
    Wt[(size_t)n * KP + k] = (__bf16)v;
}

__global__ void zero_f32_kernel(float* __restrict__ p, int n) {
    int i = blockIdx.x * blockDim.x + threadIdx.x;
    if (i < n) p[i] = 0.0f;
}

__global__ void copy_f32_kernel(const float* __restrict__ s, float* __restrict__ d, int n) {
    int i = blockIdx.x * blockDim.x + threadIdx.x;
    if (i < n) d[i] = s[i];
}

// -------------------- edge kernel: per-wave 16-edge tile through the edge MLP chain ----------------
__global__ __launch_bounds__(256, 1) void egnn_edge_kernel(
    const float*     __restrict__ h,
    const long long* __restrict__ ei,       // [2,E] int64
    const float*     __restrict__ ea,       // [E,51]
    const float*     __restrict__ pos,      // [N,3]
    const float*     __restrict__ b_e1,
    const float*     __restrict__ b_e2,
    const float*     __restrict__ b_c1,
    const float*     __restrict__ w_c2,     // [128]
    const __bf16*    __restrict__ Wt_e1,    // [128,320]
    const __bf16*    __restrict__ Wt_e2,    // [128,128]
    const __bf16*    __restrict__ Wt_c1,    // [128,128]
    float*           __restrict__ agg,      // [N,128] atomic accum
    float*           __restrict__ pos_out)  // [N,3]  atomic accum (pre-init to pos)
{
    __shared__ __bf16 actS[8][16][136];   // per-wave 16x128 activation tile (stride 136 keeps 16B align)
    __shared__ float  diffS[8][16][3];
    __shared__ int    rowS[8][16];
    __shared__ int    colS[8][16];
    __shared__ float  radS[8][16];

    const int lane = threadIdx.x & 31;
    const int w    = threadIdx.x >> 5;
    const int c    = lane & 15;      // column within 16-wide N tile (also A-row index)
    const int kh   = lane >> 4;      // K-half selector for A/B fragments
    const int tile = blockIdx.x * 8 + w;
    const int m0   = tile * 16;

    if (lane < 16) {
        int m = lane;
        int eidx = m0 + m;
        int r  = (int)ei[eidx];
        int cl = (int)ei[N_EDGES + eidx];
        rowS[w][m] = r;
        colS[w][m] = cl;
        float dx = pos[r * 3 + 0] - pos[cl * 3 + 0];
        float dy = pos[r * 3 + 1] - pos[cl * 3 + 1];
        float dz = pos[r * 3 + 2] - pos[cl * 3 + 2];
        diffS[w][m][0] = dx; diffS[w][m][1] = dy; diffS[w][m][2] = dz;
        radS[w][m] = sqrtf(dx * dx + dy * dy + dz * dz) + EPS_R;
    }
    __syncthreads();

    const int m = c;   // this lane's A-matrix row (edge within tile)
    const float* hrow = &h[(size_t)rowS[w][m] * DH];
    const float* hcol = &h[(size_t)colS[w][m] * DH];

    v8f zero = {0.f, 0.f, 0.f, 0.f, 0.f, 0.f, 0.f, 0.f};
    v8f acc[8];

    // ---------------- layer 1: [16,308p320] x [320,128] ----------------
    #pragma unroll
    for (int i = 0; i < 8; ++i) acc[i] = zero;

    for (int kc = 0; kc < 10; ++kc) {
        v16bf a;
        if (kc < 8) {
            const float* src = (kc < 4) ? (hrow + kc * 32) : (hcol + (kc - 4) * 32);
            f32x4 x0 = *(const f32x4*)(src + kh * 16);
            f32x4 x1 = *(const f32x4*)(src + kh * 16 + 4);
            f32x4 y0 = *(const f32x4*)(src + kh * 16 + 8);
            f32x4 y1 = *(const f32x4*)(src + kh * 16 + 12);
            a[0] = (__bf16)x0[0]; a[1] = (__bf16)x0[1]; a[2] = (__bf16)x0[2]; a[3] = (__bf16)x0[3];
            a[4] = (__bf16)x1[0]; a[5] = (__bf16)x1[1]; a[6] = (__bf16)x1[2]; a[7] = (__bf16)x1[3];
            a[8]  = (__bf16)y0[0]; a[9]  = (__bf16)y0[1]; a[10] = (__bf16)y0[2]; a[11] = (__bf16)y0[3];
            a[12] = (__bf16)y1[0]; a[13] = (__bf16)y1[1]; a[14] = (__bf16)y1[2]; a[15] = (__bf16)y1[3];
        } else {
            int abase = (kc - 8) * 32;   // offset into the [attr(51), radial(1), pad] tail
            #pragma unroll
            for (int e2 = 0; e2 < 16; ++e2) {
                int K = abase + kh * 16 + e2;
                float v;
                if (K < ED_DIM)       v = ea[(size_t)(m0 + m) * ED_DIM + K];
                else if (K == ED_DIM) v = radS[w][m];
                else                  v = 0.0f;
                a[e2] = (__bf16)v;
            }
        }
        #pragma unroll
        for (int nt = 0; nt < 8; ++nt) {
            v16bf b = *(const v16bf*)&Wt_e1[(size_t)(nt * 16 + c) * KP_E1 + kc * 32 + kh * 16];
            acc[nt] = wmma_bf16(a, b, acc[nt]);
        }
    }

    // bias + silu -> LDS (C/D layout: row = j + 8*kh, col = c)
    #pragma unroll
    for (int nt = 0; nt < 8; ++nt) {
        float bb = b_e1[nt * 16 + c];
        #pragma unroll
        for (int j = 0; j < 8; ++j)
            actS[w][j + 8 * kh][nt * 16 + c] = (__bf16)silu_f(acc[nt][j] + bb);
    }
    __syncthreads();

    // ---------------- layer 2: [16,128] x [128,128] ----------------
    #pragma unroll
    for (int i = 0; i < 8; ++i) acc[i] = zero;
    #pragma unroll
    for (int kc = 0; kc < 4; ++kc) {
        const __bf16* arow = &actS[w][m][kc * 32];
        v8bf r0 = *(const v8bf*)(arow + kh * 16);
        v8bf r1 = *(const v8bf*)(arow + kh * 16 + 8);
        v16bf a;
        #pragma unroll
        for (int e2 = 0; e2 < 8; ++e2) { a[e2] = r0[e2]; a[e2 + 8] = r1[e2]; }
        #pragma unroll
        for (int nt = 0; nt < 8; ++nt) {
            v16bf b = *(const v16bf*)&Wt_e2[(size_t)(nt * 16 + c) * 128 + kc * 32 + kh * 16];
            acc[nt] = wmma_bf16(a, b, acc[nt]);
        }
    }
    __syncthreads();

    // e = silu(. + b_e2): scatter to agg (atomics) and stash in LDS for coord path
    #pragma unroll
    for (int nt = 0; nt < 8; ++nt) {
        float bb = b_e2[nt * 16 + c];
        #pragma unroll
        for (int j = 0; j < 8; ++j) {
            float v = silu_f(acc[nt][j] + bb);
            int mm = j + 8 * kh;
            actS[w][mm][nt * 16 + c] = (__bf16)v;
            atomicAdd(&agg[(size_t)rowS[w][mm] * DH + nt * 16 + c], v);
        }
    }
    __syncthreads();

    // ---------------- coord path: silu(e @ W_c1 + b_c1) @ W_c2 ----------------
    #pragma unroll
    for (int i = 0; i < 8; ++i) acc[i] = zero;
    #pragma unroll
    for (int kc = 0; kc < 4; ++kc) {
        const __bf16* arow = &actS[w][m][kc * 32];
        v8bf r0 = *(const v8bf*)(arow + kh * 16);
        v8bf r1 = *(const v8bf*)(arow + kh * 16 + 8);
        v16bf a;
        #pragma unroll
        for (int e2 = 0; e2 < 8; ++e2) { a[e2] = r0[e2]; a[e2 + 8] = r1[e2]; }
        #pragma unroll
        for (int nt = 0; nt < 8; ++nt) {
            v16bf b = *(const v16bf*)&Wt_c1[(size_t)(nt * 16 + c) * 128 + kc * 32 + kh * 16];
            acc[nt] = wmma_bf16(a, b, acc[nt]);
        }
    }

    float part[8];
    #pragma unroll
    for (int j = 0; j < 8; ++j) part[j] = 0.0f;
    #pragma unroll
    for (int nt = 0; nt < 8; ++nt) {
        float wv = w_c2[nt * 16 + c];
        float bb = b_c1[nt * 16 + c];
        #pragma unroll
        for (int j = 0; j < 8; ++j)
            part[j] += silu_f(acc[nt][j] + bb) * wv;
    }
    // reduce over the 16 lanes of each half-wave (masks < 16 stay within halves)
    #pragma unroll
    for (int mask = 1; mask < 16; mask <<= 1) {
        #pragma unroll
        for (int j = 0; j < 8; ++j)
            part[j] += __shfl_xor(part[j], mask, 32);
    }
    if (c == 0) {
        #pragma unroll
        for (int j = 0; j < 8; ++j) {
            int mm = j + 8 * kh;
            float cu = fminf(1.0f, fmaxf(-1.0f, part[j]));
            int r = rowS[w][mm];
            atomicAdd(&pos_out[r * 3 + 0], cu * diffS[w][mm][0]);
            atomicAdd(&pos_out[r * 3 + 1], cu * diffS[w][mm][1]);
            atomicAdd(&pos_out[r * 3 + 2], cu * diffS[w][mm][2]);
        }
    }
}

// -------------------- node kernel: silu(concat(h,agg) @ W_n1 + b_n1) @ W_n2 + b_n2 + h -------------
__global__ __launch_bounds__(256, 1) void egnn_node_kernel(
    const float*  __restrict__ h,
    const float*  __restrict__ agg,
    const float*  __restrict__ b_n1,
    const float*  __restrict__ b_n2,
    const __bf16* __restrict__ Wt_n1,   // [128,256]
    const __bf16* __restrict__ Wt_n2,   // [128,128]
    float*        __restrict__ h_out)
{
    __shared__ __bf16 actS[8][16][136];

    const int lane = threadIdx.x & 31;
    const int w    = threadIdx.x >> 5;
    const int c    = lane & 15;
    const int kh   = lane >> 4;
    const int tile = blockIdx.x * 8 + w;
    const bool active = (tile * 16) < N_NODES;   // N_NODES % 16 == 0
    const int m0 = tile * 16;
    const int node_m = m0 + c;

    v8f zero = {0.f, 0.f, 0.f, 0.f, 0.f, 0.f, 0.f, 0.f};
    v8f acc[8];

    if (active) {
        #pragma unroll
        for (int i = 0; i < 8; ++i) acc[i] = zero;
        #pragma unroll
        for (int kc = 0; kc < 8; ++kc) {
            const float* src = (kc < 4) ? &h[(size_t)node_m * DH + kc * 32]
                                        : &agg[(size_t)node_m * DH + (kc - 4) * 32];
            f32x4 x0 = *(const f32x4*)(src + kh * 16);
            f32x4 x1 = *(const f32x4*)(src + kh * 16 + 4);
            f32x4 y0 = *(const f32x4*)(src + kh * 16 + 8);
            f32x4 y1 = *(const f32x4*)(src + kh * 16 + 12);
            v16bf a;
            a[0] = (__bf16)x0[0]; a[1] = (__bf16)x0[1]; a[2] = (__bf16)x0[2]; a[3] = (__bf16)x0[3];
            a[4] = (__bf16)x1[0]; a[5] = (__bf16)x1[1]; a[6] = (__bf16)x1[2]; a[7] = (__bf16)x1[3];
            a[8]  = (__bf16)y0[0]; a[9]  = (__bf16)y0[1]; a[10] = (__bf16)y0[2]; a[11] = (__bf16)y0[3];
            a[12] = (__bf16)y1[0]; a[13] = (__bf16)y1[1]; a[14] = (__bf16)y1[2]; a[15] = (__bf16)y1[3];
            #pragma unroll
            for (int nt = 0; nt < 8; ++nt) {
                v16bf b = *(const v16bf*)&Wt_n1[(size_t)(nt * 16 + c) * KP_N1 + kc * 32 + kh * 16];
                acc[nt] = wmma_bf16(a, b, acc[nt]);
            }
        }
        #pragma unroll
        for (int nt = 0; nt < 8; ++nt) {
            float bb = b_n1[nt * 16 + c];
            #pragma unroll
            for (int j = 0; j < 8; ++j)
                actS[w][j + 8 * kh][nt * 16 + c] = (__bf16)silu_f(acc[nt][j] + bb);
        }
    }
    __syncthreads();
    if (active) {
        #pragma unroll
        for (int i = 0; i < 8; ++i) acc[i] = zero;
        #pragma unroll
        for (int kc = 0; kc < 4; ++kc) {
            const __bf16* arow = &actS[w][c][kc * 32];
            v8bf r0 = *(const v8bf*)(arow + kh * 16);
            v8bf r1 = *(const v8bf*)(arow + kh * 16 + 8);
            v16bf a;
            #pragma unroll
            for (int e2 = 0; e2 < 8; ++e2) { a[e2] = r0[e2]; a[e2 + 8] = r1[e2]; }
            #pragma unroll
            for (int nt = 0; nt < 8; ++nt) {
                v16bf b = *(const v16bf*)&Wt_n2[(size_t)(nt * 16 + c) * 128 + kc * 32 + kh * 16];
                acc[nt] = wmma_bf16(a, b, acc[nt]);
            }
        }
        #pragma unroll
        for (int nt = 0; nt < 8; ++nt) {
            float bb = b_n2[nt * 16 + c];
            #pragma unroll
            for (int j = 0; j < 8; ++j) {
                int node = m0 + j + 8 * kh;
                size_t o = (size_t)node * DH + nt * 16 + c;
                h_out[o] = acc[nt][j] + bb + h[o];
            }
        }
    }
}

extern "C" void kernel_launch(void* const* d_in, const int* in_sizes, int n_in,
                              void* d_out, int out_size, void* d_ws, size_t ws_size,
                              hipStream_t stream) {
    const float*     h    = (const float*)d_in[0];
    const long long* ei   = (const long long*)d_in[1];   // int64 per reference
    const float*     ea   = (const float*)d_in[2];
    const float*     pos  = (const float*)d_in[3];
    const float*     W_e1 = (const float*)d_in[4];
    const float*     b_e1 = (const float*)d_in[5];
    const float*     W_e2 = (const float*)d_in[6];
    const float*     b_e2 = (const float*)d_in[7];
    const float*     W_c1 = (const float*)d_in[8];
    const float*     b_c1 = (const float*)d_in[9];
    const float*     W_c2 = (const float*)d_in[10];
    const float*     W_n1 = (const float*)d_in[11];
    const float*     b_n1 = (const float*)d_in[12];
    const float*     W_n2 = (const float*)d_in[13];
    const float*     b_n2 = (const float*)d_in[14];

    float* h_out   = (float*)d_out;
    float* pos_out = (float*)d_out + (size_t)N_NODES * DH;

    char*  ws    = (char*)d_ws;
    float* agg   = (float*)ws;
    size_t off   = (size_t)N_NODES * DH * sizeof(float);     // 25.6 MB, 32B aligned
    __bf16* Wt_e1 = (__bf16*)(ws + off); off += (size_t)DH * KP_E1 * sizeof(__bf16);
    __bf16* Wt_e2 = (__bf16*)(ws + off); off += (size_t)DH * 128   * sizeof(__bf16);
    __bf16* Wt_c1 = (__bf16*)(ws + off); off += (size_t)DH * 128   * sizeof(__bf16);
    __bf16* Wt_n1 = (__bf16*)(ws + off); off += (size_t)DH * KP_N1 * sizeof(__bf16);
    __bf16* Wt_n2 = (__bf16*)(ws + off); off += (size_t)DH * 128   * sizeof(__bf16);

    // weight prep (bf16, transposed, K-padded)
    prep_wt_kernel<<<(128 * KP_E1 + 255) / 256, 256, 0, stream>>>(W_e1, Wt_e1, 308, 128, KP_E1);
    prep_wt_kernel<<<(128 * 128 + 255) / 256, 256, 0, stream>>>(W_e2, Wt_e2, 128, 128, 128);
    prep_wt_kernel<<<(128 * 128 + 255) / 256, 256, 0, stream>>>(W_c1, Wt_c1, 128, 128, 128);
    prep_wt_kernel<<<(128 * KP_N1 + 255) / 256, 256, 0, stream>>>(W_n1, Wt_n1, 256, 128, KP_N1);
    prep_wt_kernel<<<(128 * 128 + 255) / 256, 256, 0, stream>>>(W_n2, Wt_n2, 128, 128, 128);

    // accumulator init
    zero_f32_kernel<<<(N_NODES * DH + 255) / 256, 256, 0, stream>>>(agg, N_NODES * DH);
    copy_f32_kernel<<<(N_NODES * 3 + 255) / 256, 256, 0, stream>>>(pos, pos_out, N_NODES * 3);

    // edge MLP chain: 800000/16 = 50000 tiles, 8 wave-tiles per block
    egnn_edge_kernel<<<N_EDGES / (16 * 8), 256, 0, stream>>>(
        h, ei, ea, pos, b_e1, b_e2, b_c1, W_c2, Wt_e1, Wt_e2, Wt_c1, agg, pos_out);

    // node MLP: 3125 tiles
    egnn_node_kernel<<<(N_NODES / 16 + 7) / 8, 256, 0, stream>>>(
        h, agg, b_n1, b_n2, Wt_n1, Wt_n2, h_out);
}
